// CrossAttentionBlock_70325794504981
// MI455X (gfx1250) — matrix operand order repaired and checked by reading, not compiled, not run on hardware
//
#include <hip/hip_runtime.h>
#include <hip/hip_bf16.h>
#include <math.h>

// CrossAttentionBlock for MI455X (gfx1250, wave32, WMMA).
// All GEMMs use v_wmma_f32_16x16x32_bf16; fused scores+softmax uses a 64KB LDS
// tile; everything launches on `stream` (graph-capture safe). Needs ~177MB ws.

typedef __attribute__((ext_vector_type(16))) __bf16 bf16x16;
typedef __attribute__((ext_vector_type(8)))  __bf16 bf16x8;
typedef __attribute__((ext_vector_type(8)))  float  f32x8;

#define NB 32
#define NC 512
#define NS 1024
#define ND 768
#define NL 128
#define NH 8
#define NCH 64

__device__ __forceinline__ unsigned short f2bf(float f) {
  unsigned int u = __float_as_uint(f);
  u += 0x7FFFu + ((u >> 16) & 1u);   // round-to-nearest-even
  return (unsigned short)(u >> 16);
}

// A fragment (16x32, row-major rows): per ISA table,
//   lanes 0..15:  row = lane,    K = k0..k0+7  and k0+16..k0+23
//   lanes 16..31: row = lane-16, K = k0+8..+15 and k0+24..+31
// => two 16-byte contiguous loads per lane.
__device__ __forceinline__ bf16x16 load_frag_A(const unsigned short* base, int ld,
                                               int row, int k0, int kh) {
  const __bf16* p = reinterpret_cast<const __bf16*>(base) + (size_t)row * ld + k0 + kh * 8;
  bf16x8 lo = *reinterpret_cast<const bf16x8*>(p);
  bf16x8 hi = *reinterpret_cast<const bf16x8*>(p + 16);
  bf16x16 r;
#pragma unroll
  for (int i = 0; i < 8; ++i) { r[i] = lo[i]; r[i + 8] = hi[i]; }
  return r;
}

// B fragment (32x16, K-contiguous per column since operand is stored NxK):
// column = lane&15; lanes 0..15 hold K=k0..k0+7 and k0+16..k0+23,
// lanes 16..31 hold K=k0+8..+15 and k0+24..+31 (same order as A).
__device__ __forceinline__ bf16x16 load_frag_B(const unsigned short* base, int ld,
                                               int col, int k0, int kh) {
  const __bf16* p = reinterpret_cast<const __bf16*>(base) + (size_t)col * ld + k0 + kh * 8;
  bf16x8 lo = *reinterpret_cast<const bf16x8*>(p);
  bf16x8 hi = *reinterpret_cast<const bf16x8*>(p + 16);
  bf16x16 r;
#pragma unroll
  for (int i = 0; i < 8; ++i) { r[i] = lo[i]; r[i + 8] = hi[i]; }
  return r;
}

#define WMMA_BF16(A, Bf, Cacc) \
  __builtin_amdgcn_wmma_f32_16x16x32_bf16(false, (A), false, (Bf), (short)0, (Cacc), false, false)

// 32x32 wave tile: D = W(MxK) * Xt(NxK)^T ; both operands K-contiguous.
__device__ __forceinline__ void gemm_tile_32x32(const unsigned short* W,
                                                const unsigned short* Xt,
                                                int ldW, int ldX, int K,
                                                int m0, int n0, int lane,
                                                f32x8 acc[2][2]) {
  const int r16 = lane & 15;
  const int kh  = (lane >> 4) & 1;
  for (int k0 = 0; k0 < K; k0 += 32) {
    bf16x16 a0 = load_frag_A(W,  ldW, m0 + r16,      k0, kh);
    bf16x16 a1 = load_frag_A(W,  ldW, m0 + 16 + r16, k0, kh);
    bf16x16 b0 = load_frag_B(Xt, ldX, n0 + r16,      k0, kh);
    bf16x16 b1 = load_frag_B(Xt, ldX, n0 + 16 + r16, k0, kh);
    acc[0][0] = WMMA_BF16(a0, b0, acc[0][0]);
    acc[0][1] = WMMA_BF16(a0, b1, acc[0][1]);
    acc[1][0] = WMMA_BF16(a1, b0, acc[1][0]);
    acc[1][1] = WMMA_BF16(a1, b1, acc[1][1]);
  }
}

// ---------------- conversion ----------------
__global__ void __launch_bounds__(256) f32_to_bf16_kernel(const float* __restrict__ in,
                                                          unsigned short* __restrict__ out, int n) {
  int i = blockIdx.x * 256 + threadIdx.x;
  if (i < n) out[i] = f2bf(in[i]);
}

// ---------------- GroupNorm over x: (B,C,S) -> bf16 transposed (B,S,C) ----------------
__global__ void __launch_bounds__(256) gn_x_kernel(const float* __restrict__ x,
                                                   const float* __restrict__ gw,
                                                   const float* __restrict__ gb,
                                                   unsigned short* __restrict__ out) {
  const int blk = blockIdx.x;                 // B*32 blocks
  const int b = blk >> 5, g = blk & 31;       // 16 channels x 1024 spatial per group
  const float* xb = x + ((size_t)b * NC + (size_t)g * 16) * NS;
  float s = 0.f, sq = 0.f;
  for (int i = threadIdx.x; i < 16 * NS; i += 256) { float v = xb[i]; s += v; sq += v * v; }
  __shared__ float r1[256], r2[256];
  r1[threadIdx.x] = s; r2[threadIdx.x] = sq;
  __syncthreads();
  for (int st = 128; st > 0; st >>= 1) {
    if (threadIdx.x < st) { r1[threadIdx.x] += r1[threadIdx.x + st]; r2[threadIdx.x] += r2[threadIdx.x + st]; }
    __syncthreads();
  }
  const float mu = r1[0] * (1.f / 16384.f);
  const float var = r2[0] * (1.f / 16384.f) - mu * mu;
  const float rs = rsqrtf(var + 1e-5f);
  for (int i = threadIdx.x; i < 16 * NS; i += 256) {
    int c = g * 16 + (i >> 10), sp = i & 1023;
    float v = (xb[i] - mu) * rs * gw[c] + gb[c];
    out[((size_t)b * NS + sp) * NC + c] = f2bf(v);
  }
}

// ---------------- GroupNorm over ctx channels: in/out layout (B,L,768) bf16 ----------------
__global__ void __launch_bounds__(256) gn_ctx_kernel(const float* __restrict__ ctx,
                                                     const float* __restrict__ gw,
                                                     const float* __restrict__ gb,
                                                     unsigned short* __restrict__ out) {
  const int blk = blockIdx.x;
  const int b = blk >> 5, g = blk & 31;       // 24 channels x 128 tokens per group
  float s = 0.f, sq = 0.f;
  for (int i = threadIdx.x; i < 24 * NL; i += 256) {
    int d = g * 24 + i / NL, l = i & (NL - 1);
    float v = ctx[((size_t)b * NL + l) * ND + d];
    s += v; sq += v * v;
  }
  __shared__ float r1[256], r2[256];
  r1[threadIdx.x] = s; r2[threadIdx.x] = sq;
  __syncthreads();
  for (int st = 128; st > 0; st >>= 1) {
    if (threadIdx.x < st) { r1[threadIdx.x] += r1[threadIdx.x + st]; r2[threadIdx.x] += r2[threadIdx.x + st]; }
    __syncthreads();
  }
  const float mu = r1[0] * (1.f / 3072.f);
  const float var = r2[0] * (1.f / 3072.f) - mu * mu;
  const float rs = rsqrtf(var + 1e-5f);
  for (int i = threadIdx.x; i < 24 * NL; i += 256) {
    int d = g * 24 + i / NL, l = i & (NL - 1);
    float v = (ctx[((size_t)b * NL + l) * ND + d] - mu) * rs * gw[d] + gb[d];
    out[((size_t)b * NL + l) * ND + d] = f2bf(v);
  }
}

// ---------------- q = q_w * gn(x) + q_b ; stored transposed (B,S,C) bf16 ----------------
__global__ void __launch_bounds__(256) gemm_q_kernel(const unsigned short* __restrict__ qw,
                                                     const unsigned short* __restrict__ gnx,
                                                     const float* __restrict__ qb,
                                                     unsigned short* __restrict__ qt) {
  const int lane = threadIdx.x & 31, wid = threadIdx.x >> 5;
  const int b = blockIdx.z;
  const int m0 = blockIdx.y * 128 + (wid >> 1) * 32;
  const int n0 = blockIdx.x * 64 + (wid & 1) * 32;
  const unsigned short* Xt = gnx + (size_t)b * NS * NC;
  f32x8 acc[2][2] = {};
  gemm_tile_32x32(qw, Xt, NC, NC, NC, m0, n0, lane, acc);
  const int r16 = lane & 15, kh = (lane >> 4) & 1;
  unsigned short* qtb = qt + (size_t)b * NS * NC;
#pragma unroll
  for (int i = 0; i < 2; ++i)
#pragma unroll
    for (int j = 0; j < 2; ++j)
#pragma unroll
      for (int r = 0; r < 8; ++r) {
        int m = m0 + i * 16 + r + kh * 8;  // output channel
        int n = n0 + j * 16 + r16;         // spatial position
        qtb[(size_t)n * NC + m] = f2bf(acc[i][j][r] + qb[m]);
      }
}

// ---------------- kv = kv_w * gn(ctx) + kv_b ; k -> (B,L,C), v -> (B,C,L) ----------------
__global__ void __launch_bounds__(256) gemm_kv_kernel(const unsigned short* __restrict__ kvw,
                                                      const unsigned short* __restrict__ gnc,
                                                      const float* __restrict__ kvb,
                                                      unsigned short* __restrict__ kt,
                                                      unsigned short* __restrict__ vv) {
  const int lane = threadIdx.x & 31, wid = threadIdx.x >> 5;
  const int b = blockIdx.z;
  const int m0 = blockIdx.y * 128 + (wid >> 1) * 32;   // oc in [0,1024)
  const int n0 = blockIdx.x * 64 + (wid & 1) * 32;     // l  in [0,128)
  const unsigned short* Xt = gnc + (size_t)b * NL * ND;
  f32x8 acc[2][2] = {};
  gemm_tile_32x32(kvw, Xt, ND, ND, ND, m0, n0, lane, acc);
  const int r16 = lane & 15, kh = (lane >> 4) & 1;
#pragma unroll
  for (int i = 0; i < 2; ++i)
#pragma unroll
    for (int j = 0; j < 2; ++j)
#pragma unroll
      for (int r = 0; r < 8; ++r) {
        int m = m0 + i * 16 + r + kh * 8;
        int n = n0 + j * 16 + r16;
        float val = acc[i][j][r] + kvb[m];
        if (m < NC) kt[(size_t)b * NL * NC + (size_t)n * NC + m] = f2bf(val);
        else        vv[(size_t)b * NC * NL + (size_t)(m - NC) * NL + n] = f2bf(val);
      }
}

// ---------------- fused scores (Q^T K / 8) + mask + softmax -> probs bf16 (z,S,L) ----------------
__global__ void __launch_bounds__(256) scores_softmax_kernel(const unsigned short* __restrict__ qt,
                                                             const unsigned short* __restrict__ kt,
                                                             const int* __restrict__ mask,
                                                             unsigned short* __restrict__ probs) {
  __shared__ float sm[128 * 128];                       // 64KB score tile
  const int lane = threadIdx.x & 31, wid = threadIdx.x >> 5;
  const int z = blockIdx.y, b = z >> 3, h = z & 7;
  const int sbase = blockIdx.x * 128;
  const int m0 = (wid >> 1) * 32;                       // local s tile (4 waves)
  const int n0 = (wid & 1) * 64;                        // local l tile (2 waves)
  const unsigned short* W = qt + (size_t)b * NS * NC + h * NCH;   // rows=s, ld=NC, K=64
  const unsigned short* X = kt + (size_t)b * NL * NC + h * NCH;   // rows=l, ld=NC
  const int r16 = lane & 15, kh = (lane >> 4) & 1;
  f32x8 acc[2][4] = {};
  for (int k0 = 0; k0 < NCH; k0 += 32) {
    bf16x16 a0 = load_frag_A(W, NC, sbase + m0 + r16,      k0, kh);
    bf16x16 a1 = load_frag_A(W, NC, sbase + m0 + 16 + r16, k0, kh);
#pragma unroll
    for (int j = 0; j < 4; ++j) {
      bf16x16 bj = load_frag_B(X, NC, n0 + j * 16 + r16, k0, kh);
      acc[0][j] = WMMA_BF16(a0, bj, acc[0][j]);
      acc[1][j] = WMMA_BF16(a1, bj, acc[1][j]);
    }
  }
  const int* mb = mask + b * NL;
#pragma unroll
  for (int i = 0; i < 2; ++i)
#pragma unroll
    for (int j = 0; j < 4; ++j)
#pragma unroll
      for (int r = 0; r < 8; ++r) {
        int row = m0 + i * 16 + r + kh * 8;
        int col = n0 + j * 16 + r16;
        sm[row * 128 + col] = (mb[col] != 0) ? acc[i][j][r] * 0.125f : -3.402823466e38f;
      }
  __syncthreads();
  // softmax: each wave owns 16 rows; 4 elements/lane; shfl reductions (wave32)
  unsigned short* pz = probs + (size_t)z * NS * NL + (size_t)sbase * NL;
  for (int rr = 0; rr < 16; ++rr) {
    int row = wid * 16 + rr;
    float v[4];
#pragma unroll
    for (int e = 0; e < 4; ++e) v[e] = sm[row * 128 + e * 32 + lane];
    float mx = fmaxf(fmaxf(v[0], v[1]), fmaxf(v[2], v[3]));
    for (int off = 16; off > 0; off >>= 1) mx = fmaxf(mx, __shfl_xor(mx, off, 32));
    float p[4], s = 0.f;
#pragma unroll
    for (int e = 0; e < 4; ++e) { p[e] = __expf(v[e] - mx); s += p[e]; }
    for (int off = 16; off > 0; off >>= 1) s += __shfl_xor(s, off, 32);
    float inv = 1.0f / s;
#pragma unroll
    for (int e = 0; e < 4; ++e) pz[(size_t)row * NL + e * 32 + lane] = f2bf(p[e] * inv);
  }
}

// ---------------- out = P * V^T per head ; stored transposed (B,S,C) bf16 ----------------
__global__ void __launch_bounds__(256) gemm_attn_kernel(const unsigned short* __restrict__ probs,
                                                        const unsigned short* __restrict__ vv,
                                                        unsigned short* __restrict__ outt) {
  const int lane = threadIdx.x & 31, wid = threadIdx.x >> 5;
  const int z = blockIdx.z, b = z >> 3, h = z & 7;
  const int m0 = blockIdx.y * 128 + (wid >> 1) * 32;   // s
  const int n0 = (wid & 1) * 32;                       // c within head [0,64)
  const unsigned short* W = probs + (size_t)z * NS * NL;                       // rows=s, ld=NL, K=128
  const unsigned short* X = vv + (size_t)b * NC * NL + (size_t)h * NCH * NL;   // rows=c_local, ld=NL
  f32x8 acc[2][2] = {};
  gemm_tile_32x32(W, X, NL, NL, NL, m0, n0, lane, acc);
  const int r16 = lane & 15, kh = (lane >> 4) & 1;
#pragma unroll
  for (int i = 0; i < 2; ++i)
#pragma unroll
    for (int j = 0; j < 2; ++j)
#pragma unroll
      for (int r = 0; r < 8; ++r) {
        int m = m0 + i * 16 + r + kh * 8;  // s
        int n = n0 + j * 16 + r16;         // c local
        outt[(size_t)b * NS * NC + (size_t)m * NC + h * NCH + n] = f2bf(acc[i][j][r]);
      }
}

// ---------------- y = proj_w * out + proj_b + x ; fp32 (B,C,S) ----------------
__global__ void __launch_bounds__(256) gemm_proj_kernel(const unsigned short* __restrict__ pw,
                                                        const unsigned short* __restrict__ outt,
                                                        const float* __restrict__ pb,
                                                        const float* __restrict__ x,
                                                        float* __restrict__ y) {
  const int lane = threadIdx.x & 31, wid = threadIdx.x >> 5;
  const int b = blockIdx.z;
  const int m0 = blockIdx.y * 128 + (wid >> 1) * 32;   // c out
  const int n0 = blockIdx.x * 64 + (wid & 1) * 32;     // s
  const unsigned short* Xt = outt + (size_t)b * NS * NC;
  f32x8 acc[2][2] = {};
  gemm_tile_32x32(pw, Xt, NC, NC, NC, m0, n0, lane, acc);
  const int r16 = lane & 15, kh = (lane >> 4) & 1;
#pragma unroll
  for (int i = 0; i < 2; ++i)
#pragma unroll
    for (int j = 0; j < 2; ++j)
#pragma unroll
      for (int r = 0; r < 8; ++r) {
        int m = m0 + i * 16 + r + kh * 8;
        int n = n0 + j * 16 + r16;
        size_t idx = (size_t)b * NC * NS + (size_t)m * NS + n;
        y[idx] = acc[i][j][r] + pb[m] + x[idx];
      }
}

extern "C" void kernel_launch(void* const* d_in, const int* in_sizes, int n_in,
                              void* d_out, int out_size, void* d_ws, size_t ws_size,
                              hipStream_t stream) {
  (void)in_sizes; (void)n_in; (void)out_size; (void)ws_size;
  const float* x        = (const float*)d_in[0];
  const float* context  = (const float*)d_in[1];
  const int*   mask     = (const int*)d_in[2];
  const float* norm_w   = (const float*)d_in[3];
  const float* norm_b   = (const float*)d_in[4];
  const float* normc_w  = (const float*)d_in[5];
  const float* normc_b  = (const float*)d_in[6];
  const float* q_w      = (const float*)d_in[7];
  const float* q_b      = (const float*)d_in[8];
  const float* kv_w     = (const float*)d_in[9];
  const float* kv_b     = (const float*)d_in[10];
  const float* proj_w   = (const float*)d_in[11];
  const float* proj_b   = (const float*)d_in[12];
  float* out            = (float*)d_out;

  char* ws = (char*)d_ws;
  size_t off = 0;
  auto take = [&](size_t bytes) -> void* {
    void* p = ws + off;
    off += (bytes + 255) & ~(size_t)255;
    return p;
  };
  unsigned short* qw_bf  = (unsigned short*)take((size_t)NC * NC * 2);        // 0.5 MB
  unsigned short* kvw_bf = (unsigned short*)take((size_t)2 * NC * ND * 2);    // 1.5 MB
  unsigned short* pw_bf  = (unsigned short*)take((size_t)NC * NC * 2);        // 0.5 MB
  unsigned short* gnx    = (unsigned short*)take((size_t)NB * NS * NC * 2);   // 32 MB (B,S,C)
  unsigned short* gnc    = (unsigned short*)take((size_t)NB * NL * ND * 2);   // 6 MB  (B,L,D)
  unsigned short* qt     = (unsigned short*)take((size_t)NB * NS * NC * 2);   // 32 MB (B,S,C)
  unsigned short* kt     = (unsigned short*)take((size_t)NB * NL * NC * 2);   // 4 MB  (B,L,C)
  unsigned short* vv     = (unsigned short*)take((size_t)NB * NC * NL * 2);   // 4 MB  (B,C,L)
  unsigned short* probs  = (unsigned short*)take((size_t)NB * NH * NS * NL * 2); // 64 MB
  unsigned short* outt   = (unsigned short*)take((size_t)NB * NS * NC * 2);   // 32 MB (B,S,C)
  // total ~ 177 MB

  // 1) weight conversion
  f32_to_bf16_kernel<<<(NC * NC + 255) / 256, 256, 0, stream>>>(q_w, qw_bf, NC * NC);
  f32_to_bf16_kernel<<<(2 * NC * ND + 255) / 256, 256, 0, stream>>>(kv_w, kvw_bf, 2 * NC * ND);
  f32_to_bf16_kernel<<<(NC * NC + 255) / 256, 256, 0, stream>>>(proj_w, pw_bf, NC * NC);
  // 2) group norms
  gn_x_kernel<<<NB * 32, 256, 0, stream>>>(x, norm_w, norm_b, gnx);
  gn_ctx_kernel<<<NB * 32, 256, 0, stream>>>(context, normc_w, normc_b, gnc);
  // 3) projections
  gemm_q_kernel<<<dim3(NS / 64, NC / 128, NB), 256, 0, stream>>>(qw_bf, gnx, q_b, qt);
  gemm_kv_kernel<<<dim3(NL / 64, 2 * NC / 128, NB), 256, 0, stream>>>(kvw_bf, gnc, kv_b, kt, vv);
  // 4) fused scores + mask + softmax (grid.x = S tiles, grid.y = b*h)
  scores_softmax_kernel<<<dim3(NS / 128, NB * NH), 256, 0, stream>>>(qt, kt, mask, probs);
  // 5) attention output
  gemm_attn_kernel<<<dim3(1, NS / 128, NB * NH), 256, 0, stream>>>(probs, vv, outt);
  // 6) final projection + residual
  gemm_proj_kernel<<<dim3(NS / 64, NC / 128, NB), 256, 0, stream>>>(pw_bf, outt, proj_b, x, out);
}
